// AdditiveAttention_17360257810637
// MI455X (gfx1250) — matrix operand order repaired and checked by reading, not compiled, not run on hardware
//
#include <hip/hip_runtime.h>
#include <hip/hip_bf16.h>

typedef __attribute__((ext_vector_type(16))) _Float16 v16h;
typedef __attribute__((ext_vector_type(8)))  _Float16 v8h;
typedef __attribute__((ext_vector_type(8)))  float    v8f;
typedef __attribute__((ext_vector_type(4)))  unsigned int u32x4;
typedef __attribute__((ext_vector_type(4)))  int      i32x4;
typedef __attribute__((ext_vector_type(8)))  int      i32x8;

#define NEG_INF_VAL (-1000000.0f)

// Problem sizes (fixed by the reference)
#define BB   8
#define LQ   256
#define LK   1024
#define QS   256
#define KS   256
#define HH   128
#define VS   256

#define BT_STRIDE 1032   // halfs; 2064B row stride => 4-bank step, conflict-free b128 reads

#if defined(__has_builtin)
#if __has_builtin(__builtin_amdgcn_tensor_load_to_lds) && \
    __has_builtin(__builtin_amdgcn_s_wait_tensorcnt)   && \
    __has_builtin(__builtin_amdgcn_readfirstlane)
#define HAVE_TDM 1
#endif
#endif

// ---------------------------------------------------------------------------
// fp32 -> fp16 elementwise convert
// ---------------------------------------------------------------------------
__global__ void cvt_f32_to_f16(const float* __restrict__ in,
                               _Float16* __restrict__ out, int n) {
    int i = blockIdx.x * blockDim.x + threadIdx.x;
    if (i < n) out[i] = (_Float16)in[i];
}

#if defined(HAVE_TDM)
__device__ __forceinline__ unsigned int rfl_u32(unsigned int x) {
    return (unsigned int)__builtin_amdgcn_readfirstlane((int)x);
}

// Issue one TDM transfer: 16 rows x 32 K-elements (f16) of A into LDS.
// Descriptor per CDNA5 ISA ch.8: data_size=2B, tile 32x16, dim0_stride=K,
// tensor dims large (no OOB), type=2, group2/3 zero (2D tensor).
__device__ __forceinline__ void tdm_load_a_slab(const _Float16* gsrc,
                                                unsigned int ldsA, int K) {
    const unsigned long long gA = (unsigned long long)(const void*)gsrc;

    u32x4 g0;
    g0[0] = 1u;                                            // count=1
    g0[1] = rfl_u32(ldsA);                                 // lds_addr
    g0[2] = rfl_u32((unsigned int)(gA & 0xffffffffull));   // global_addr lo
    g0[3] = rfl_u32((unsigned int)((gA >> 32) & 0x01ffffffull))
            | 0x80000000u;                                 // addr hi | type=2

    i32x8 g1;
    g1[0] = 0x00010000;                    // workgroup_mask=0, data_size=2B
    g1[1] = 0;                             // tensor_dim0[15:0] (dim0 = 1<<24)
    g1[2] = 0x0100;                        // dim0[31:16], dim1[15:0]
    g1[3] = (32 << 16) | 0x0100;           // tile_dim0=32, dim1[31:16]
    g1[4] = 16;                            // tile_dim1=16, tile_dim2=0
    g1[5] = rfl_u32((unsigned int)K);      // tensor_dim0_stride = K elems
    g1[6] = 0;
    g1[7] = 0;

    const i32x4 z4 = {};
#if __clang_major__ >= 23
    const i32x8 z8 = {};
    __builtin_amdgcn_tensor_load_to_lds(g0, g1, z4, z4, z8, 0);
#else
    __builtin_amdgcn_tensor_load_to_lds(g0, g1, z4, z4, 0);
#endif
}
#endif

// ---------------------------------------------------------------------------
// Batched row-major GEMM: C[M,N] (f32) = A[M,K] (f16) * B[K,N] (f16)
// Block = 128 threads (4 waves), one 16x16 tile per wave; all waves share one
// 16-column B tile staged TRANSPOSED in LDS (fragments = 2x ds_load_b128).
// A slabs (16x32 f16) staged by the Tensor Data Mover, double-buffered:
// issue slab(kk+32) -> s_wait_tensorcnt 1 -> compute slab(kk).
// Fragment layouts per CDNA5 ISA 7.12.2.
// ---------------------------------------------------------------------------
__global__ void wmma_gemm_f16(const _Float16* __restrict__ A,
                              const _Float16* __restrict__ B,
                              float* __restrict__ C,
                              int M, int N, int K,
                              long long sA, long long sB, long long sC) {
    const _Float16* Ab = A + sA * (long long)blockIdx.z;
    const _Float16* Bb = B + sB * (long long)blockIdx.z;
    float*          Cb = C + sC * (long long)blockIdx.z;

    const int tid   = threadIdx.x;
    const int lane  = tid & 31;
    const int wave  = tid >> 5;
    const int tileM = blockIdx.y * 4 + wave;
    const int tileN = blockIdx.x;
    const int half  = lane >> 4;
    const int l16   = lane & 15;
    const int n0    = tileN * 16;
    const int ncol  = n0 + l16;

    __shared__ __align__(16) _Float16 Bt[16][BT_STRIDE];     // transposed B tile
#if defined(HAVE_TDM)
    __shared__ __align__(16) _Float16 Aslab[4][2][16][32];   // per-wave double buffer
    const _Float16* Abase = Ab + (long long)(tileM * 16) * K;
    const unsigned int lds0 = (unsigned int)(unsigned long long)&Aslab[wave][0][0][0];
    const unsigned int lds1 = (unsigned int)(unsigned long long)&Aslab[wave][1][0][0];
    // Prologue: start slab(kk=0) so the TDM overlaps the B staging below.
    tdm_load_a_slab(Abase, lds0, K);
#endif

    // ---- stage B tile transposed: Bt[n][k] = B[k][n0+n] ----
    for (int base = 0; base < K; base += 64) {
        const int k  = base + (tid >> 1);
        const int nh = (tid & 1) * 8;
        __builtin_prefetch(Bb + (long long)(base + 64) * N + n0 + nh, 0, 1);
        const v8h row = *(const v8h*)(Bb + (long long)k * N + n0 + nh);
#pragma unroll
        for (int j = 0; j < 8; ++j) Bt[nh + j][k] = row[j];
    }
    __syncthreads();

    const _Float16* arowg = Ab + (long long)(tileM * 16 + l16) * K;

    v8f acc = {};
    int buf = 0;
    for (int kk = 0; kk < K; kk += 32) {
        v8h alo, ahi;
#if defined(HAVE_TDM)
        // Issue the next slab into the other buffer (duplicate on last iter so
        // the immediate wait value stays constant), then wait for the current.
        int kn = kk + 32;
        if (kn >= K) kn = kk;
        tdm_load_a_slab(Abase + kn, buf ? lds0 : lds1, K);
        __builtin_amdgcn_s_wait_tensorcnt((short)1);
        asm volatile("" ::: "memory");   // TDM wrote LDS: force fragment reload

        const _Float16* arow = &Aslab[wave][buf][l16][0];
        alo = *(const v8h*)(arow + 8 * half);
        ahi = *(const v8h*)(arow + 16 + 8 * half);
        buf ^= 1;
#else
        alo = *(const v8h*)(arowg + kk + 8 * half);
        ahi = *(const v8h*)(arowg + kk + 16 + 8 * half);
#endif
        const v16h a = __builtin_shufflevector(alo, ahi,
                        0, 1, 2, 3, 4, 5, 6, 7, 8, 9, 10, 11, 12, 13, 14, 15);

        // B fragment: lane column = l16, K = kk + 8*half + 0..7, kk + 16 + 8*half + 0..7
        const int ks = kk + 8 * half;
        const v8h blo = *(const v8h*)(&Bt[l16][ks]);
        const v8h bhi = *(const v8h*)(&Bt[l16][ks + 16]);
        const v16h b = __builtin_shufflevector(blo, bhi,
                        0, 1, 2, 3, 4, 5, 6, 7, 8, 9, 10, 11, 12, 13, 14, 15);

        acc = __builtin_amdgcn_wmma_f32_16x16x32_f16(
            /*neg_a=*/false, a, /*neg_b=*/false, b,
            /*c_mod=*/(short)0, acc, /*reuse_a=*/false, /*reuse_b=*/false);
    }
    (void)arowg;
    (void)buf;

#pragma unroll
    for (int v = 0; v < 8; ++v) {
        const int row = tileM * 16 + v + 8 * half;
        Cb[(long long)row * N + ncol] = acc[v];
    }
}

// ---------------------------------------------------------------------------
// Additive-attention scores + masked softmax (268M tanh: the irreducible cost)
// One 256-thread block per (b, q) row; 4 keys per thread.
// ---------------------------------------------------------------------------
__global__ void score_softmax(const float* __restrict__ qh,   // [B*LQ, H]
                              const float* __restrict__ kh,   // [B*LK, H]
                              const float* __restrict__ Wv,   // [H]
                              const int*   __restrict__ vlen, // [B]
                              _Float16* __restrict__ attn) {  // [B, LQ, LK]
    const int q = blockIdx.x;
    const int b = blockIdx.y;
    const int t = threadIdx.x;

    __shared__ float s_q[HH];
    __shared__ float s_wv[HH];
    __shared__ float red[256];

    if (t < HH) {
        s_q[t]  = qh[(long long)(b * LQ + q) * HH + t];
        s_wv[t] = Wv[t];
    }
    __syncthreads();

    const int vl = vlen[b];
    float sc[4];
#pragma unroll
    for (int j = 0; j < 4; ++j) {
        const int k = t + j * 256;
        const float* krow = kh + (long long)(b * LK + k) * HH;
        float acc = 0.0f;
        for (int h = 0; h < HH; h += 4) {
            const float4 kv = *(const float4*)(krow + h);
            acc += s_wv[h]     * tanhf(s_q[h]     + kv.x);
            acc += s_wv[h + 1] * tanhf(s_q[h + 1] + kv.y);
            acc += s_wv[h + 2] * tanhf(s_q[h + 2] + kv.z);
            acc += s_wv[h + 3] * tanhf(s_q[h + 3] + kv.w);
        }
        sc[j] = (k < vl) ? acc : NEG_INF_VAL;
    }

    float mx = fmaxf(fmaxf(sc[0], sc[1]), fmaxf(sc[2], sc[3]));
    red[t] = mx;
    __syncthreads();
    for (int s = 128; s > 0; s >>= 1) {
        if (t < s) red[t] = fmaxf(red[t], red[t + s]);
        __syncthreads();
    }
    mx = red[0];
    __syncthreads();

    float e[4];
    float sum = 0.0f;
#pragma unroll
    for (int j = 0; j < 4; ++j) {
        e[j] = __expf(sc[j] - mx);
        sum += e[j];
    }
    red[t] = sum;
    __syncthreads();
    for (int s = 128; s > 0; s >>= 1) {
        if (t < s) red[t] += red[t + s];
        __syncthreads();
    }
    const float inv = 1.0f / red[0];

#pragma unroll
    for (int j = 0; j < 4; ++j) {
        attn[(long long)(b * LQ + q) * LK + t + j * 256] = (_Float16)(e[j] * inv);
    }
}

// ---------------------------------------------------------------------------
// Launch
// ---------------------------------------------------------------------------
extern "C" void kernel_launch(void* const* d_in, const int* in_sizes, int n_in,
                              void* d_out, int out_size, void* d_ws, size_t ws_size,
                              hipStream_t stream) {
    const float* queries = (const float*)d_in[0];  // [B, LQ, QS]
    const float* key     = (const float*)d_in[1];  // [B, LK, KS]
    const float* value   = (const float*)d_in[2];  // [B, LK, VS]
    const int*   vlen    = (const int*)  d_in[3];  // [B]
    const float* Wq      = (const float*)d_in[4];  // [QS, H]
    const float* Wk      = (const float*)d_in[5];  // [KS, H]
    const float* Wv      = (const float*)d_in[6];  // [H]
    float* out = (float*)d_out;                    // [B, LQ, VS]

    char*  ws  = (char*)d_ws;
    size_t off = 0;
    auto alloc = [&](size_t bytes) -> void* {
        void* p = ws + off;
        off += (bytes + 255) & ~(size_t)255;
        return p;
    };

    const int nQ  = BB * LQ * QS;
    const int nK  = BB * LK * KS;
    const int nV  = BB * LK * VS;
    const int nWq = QS * HH;
    const int nWk = KS * HH;
    const int nQh = BB * LQ * HH;
    const int nKh = BB * LK * HH;
    const int nAt = BB * LQ * LK;

    _Float16* q16  = (_Float16*)alloc((size_t)nQ  * 2);
    _Float16* k16  = (_Float16*)alloc((size_t)nK  * 2);
    _Float16* v16  = (_Float16*)alloc((size_t)nV  * 2);
    _Float16* wq16 = (_Float16*)alloc((size_t)nWq * 2);
    _Float16* wk16 = (_Float16*)alloc((size_t)nWk * 2);
    float*    qh   = (float*)   alloc((size_t)nQh * 4);
    float*    kh   = (float*)   alloc((size_t)nKh * 4);
    _Float16* at16 = (_Float16*)alloc((size_t)nAt * 2);

    cvt_f32_to_f16<<<(nQ  + 255) / 256, 256, 0, stream>>>(queries, q16,  nQ);
    cvt_f32_to_f16<<<(nK  + 255) / 256, 256, 0, stream>>>(key,     k16,  nK);
    cvt_f32_to_f16<<<(nV  + 255) / 256, 256, 0, stream>>>(value,   v16,  nV);
    cvt_f32_to_f16<<<(nWq + 255) / 256, 256, 0, stream>>>(Wq,      wq16, nWq);
    cvt_f32_to_f16<<<(nWk + 255) / 256, 256, 0, stream>>>(Wk,      wk16, nWk);

    // qh = queries * Wq : M=2048, N=128, K=256
    {
        dim3 grid(HH / 16, (BB * LQ) / 64, 1);
        wmma_gemm_f16<<<grid, 128, 0, stream>>>(q16, wq16, qh,
                                                BB * LQ, HH, QS, 0, 0, 0);
    }
    // kh = key * Wk : M=8192, N=128, K=256
    {
        dim3 grid(HH / 16, (BB * LK) / 64, 1);
        wmma_gemm_f16<<<grid, 128, 0, stream>>>(k16, wk16, kh,
                                                BB * LK, HH, KS, 0, 0, 0);
    }
    // scores + masked softmax
    {
        dim3 grid(LQ, BB, 1);
        score_softmax<<<grid, 256, 0, stream>>>(qh, kh, Wv, vlen, at16);
    }
    // out[b] = attn[b] * value[b] : M=256, N=256, K=1024, batched over B
    {
        dim3 grid(VS / 16, LQ / 64, BB);
        wmma_gemm_f16<<<grid, 128, 0, stream>>>(at16, v16, out,
                                                LQ, VS, LK,
                                                (long long)LQ * LK,
                                                (long long)LK * VS,
                                                (long long)LQ * VS);
    }
}
